// PrototypicalNetwork_4887672783282
// MI455X (gfx1250) — matrix operand order repaired and checked by reading, not compiled, not run on hardware
//
#include <hip/hip_runtime.h>
#include <hip/hip_bf16.h>
#include <math.h>

#define D_IN   2048
#define D_EMB  1024
#define N_SUP  2048
#define N_QRY  16384
#define N_WAY  64

typedef unsigned short u16;
typedef u16   u16x8   __attribute__((ext_vector_type(8)));
typedef u16   u16x16  __attribute__((ext_vector_type(16)));
typedef __bf16 bf16x16 __attribute__((ext_vector_type(16)));
typedef float f32x8   __attribute__((ext_vector_type(8)));

// CDNA5 async global->LDS path (guarded: falls back to synchronous copy)
#if defined(__has_builtin)
#  if __has_builtin(__builtin_amdgcn_global_load_async_to_lds_b128)
#    define HAVE_ASYNC_LDS 1
#  endif
#  if __has_builtin(__builtin_amdgcn_s_wait_asynccnt)
#    define WAIT_ASYNC() __builtin_amdgcn_s_wait_asynccnt(0)
#  else
#    define WAIT_ASYNC() asm volatile("s_wait_asynccnt 0" ::: "memory")
#  endif
#endif
#ifndef HAVE_ASYNC_LDS
#  define HAVE_ASYNC_LDS 0
#endif

#if HAVE_ASYNC_LDS
// Param 0: AS1 ("__device__") pointer to 16B int vector; param 1: AS3 ("__shared__")
typedef int v4i __attribute__((vector_size(16)));
typedef __attribute__((address_space(1))) v4i glb_v4i;
typedef __attribute__((address_space(3))) v4i lds_v4i;
#endif

// fp32 -> bf16, round-to-nearest-even
__device__ __forceinline__ u16 f2bf(float f) {
    unsigned u = __builtin_bit_cast(unsigned, f);
    u += 0x7FFFu + ((u >> 16) & 1u);
    return (u16)(u >> 16);
}
__device__ __forceinline__ float bf2f(u16 h) {
    return __builtin_bit_cast(float, (unsigned)h << 16);
}

// Assemble a 16-element bf16 WMMA fragment from two 8-element LDS loads.
__device__ __forceinline__ bf16x16 mk_frag(u16x8 lo, u16x8 hi) {
    u16x16 t;
#pragma unroll
    for (int i = 0; i < 8; ++i) { t[i] = lo[i]; t[i + 8] = hi[i]; }
    return __builtin_bit_cast(bf16x16, t);
}

__device__ __forceinline__ f32x8 wmma_bf16(bf16x16 a, bf16x16 b, f32x8 c) {
    // (neg_a, A, neg_b, B, c_mod, C, reuse_a, reuse_b)
    return __builtin_amdgcn_wmma_f32_16x16x32_bf16(false, a, false, b, (short)0, c, false, false);
}

// ---------------------------------------------------------------------------
// Kernel 1: E = X @ W + b   (X: [M, D_IN] fp32, W: [D_IN, D_EMB] fp32)
// Block tile: 128(M) x 64(N), 256 threads = 8 waves, each wave 32x32 (2x2 WMMA).
// OUT_BF16 => store E as bf16 (query path), else fp32 (support path).
// ---------------------------------------------------------------------------
template <bool OUT_BF16>
__global__ __launch_bounds__(256) void embed_kernel(const float* __restrict__ X,
                                                    const float* __restrict__ W,
                                                    const float* __restrict__ bias,
                                                    void* __restrict__ Eout) {
    constexpr int AS = 40;  // padded LDS row stride (bf16 elems) -> conflict-free
    constexpr int BS = 40;
    __shared__ __align__(16) u16 sA[128 * AS];   // A tile, [row][k]
    __shared__ __align__(16) u16 sB[64 * BS];    // B tile, [col][k] (transposed)

    const int tid  = threadIdx.x;
    const int lane = tid & 31;
    const int w    = tid >> 5;
    const int l16  = lane & 15;
    const int h16  = lane >> 4;          // 0: lanes 0-15, 1: lanes 16-31
    const int wm   = (w >> 1) << 5;      // 0,32,64,96
    const int wn   = (w & 1) << 5;       // 0,32
    const int bn0  = blockIdx.x * 64;
    const int bm0  = blockIdx.y * 128;

    f32x8 acc[2][2];
#pragma unroll
    for (int nt = 0; nt < 2; ++nt) {
        const float bv = bias[bn0 + wn + nt * 16 + l16];
#pragma unroll
        for (int mt = 0; mt < 2; ++mt)
#pragma unroll
            for (int g = 0; g < 8; ++g) acc[mt][nt][g] = bv;
    }

    for (int kk = 0; kk < D_IN; kk += 32) {
        // --- A tile: 128x32 fp32 -> bf16 in LDS ---
#pragma unroll
        for (int j = 0; j < 4; ++j) {
            const int e = j * 256 + tid;            // 0..1023 float4s
            const int r = e >> 3, c4 = (e & 7) << 2;
            const float4 v = *(const float4*)(X + (size_t)(bm0 + r) * D_IN + kk + c4);
            const int o = r * AS + c4;
            sA[o] = f2bf(v.x); sA[o + 1] = f2bf(v.y);
            sA[o + 2] = f2bf(v.z); sA[o + 3] = f2bf(v.w);
        }
        // --- B tile: 32x64 fp32 -> bf16, stored [col][k] ---
#pragma unroll
        for (int j = 0; j < 2; ++j) {
            const int e = j * 256 + tid;            // 0..511 float4s
            const int r = e >> 4, c = (e & 15) << 2;
            const float4 v = *(const float4*)(W + (size_t)(kk + r) * D_EMB + bn0 + c);
            sB[(c + 0) * BS + r] = f2bf(v.x);
            sB[(c + 1) * BS + r] = f2bf(v.y);
            sB[(c + 2) * BS + r] = f2bf(v.z);
            sB[(c + 3) * BS + r] = f2bf(v.w);
        }
        // prime L2->WGP for the next K step (global_prefetch_b8)
        if (kk + 32 < D_IN) {
            __builtin_prefetch(X + (size_t)(bm0 + (tid >> 1)) * D_IN + (kk + 32) + ((tid & 1) << 4), 0, 0);
            if (tid < 64)
                __builtin_prefetch(W + (size_t)(kk + 32 + (tid >> 1)) * D_EMB + bn0 + ((tid & 1) << 5), 0, 0);
        }
        __syncthreads();

        // --- fragments (ISA 7.12.2 layouts) ---
        bf16x16 af[2], br[2];
        const int ka = h16 << 3;   // A: lanes<16 -> K 0..7 & 16..23; else 8..15 & 24..31
#pragma unroll
        for (int mt = 0; mt < 2; ++mt) {
            const int ro = (wm + mt * 16 + l16) * AS + ka;
            af[mt] = mk_frag(*(const u16x8*)&sA[ro], *(const u16x8*)&sA[ro + 16]);
        }
        const int kb = h16 << 3;   // B: lanes<16 -> K 0..7 & 16..23; else 8..15 & 24..31
#pragma unroll
        for (int nt = 0; nt < 2; ++nt) {
            const int co = (wn + nt * 16 + l16) * BS + kb;
            br[nt] = mk_frag(*(const u16x8*)&sB[co], *(const u16x8*)&sB[co + 16]);
        }
#pragma unroll
        for (int mt = 0; mt < 2; ++mt)
#pragma unroll
            for (int nt = 0; nt < 2; ++nt)
                acc[mt][nt] = wmma_bf16(af[mt], br[nt], acc[mt][nt]);
        __syncthreads();
    }

    // --- epilogue: C layout = VGPR g -> row g (lanes 0-15) / g+8 (lanes 16-31) ---
#pragma unroll
    for (int mt = 0; mt < 2; ++mt)
#pragma unroll
        for (int nt = 0; nt < 2; ++nt) {
            const int col = bn0 + wn + nt * 16 + l16;
#pragma unroll
            for (int g = 0; g < 8; ++g) {
                const int row = bm0 + wm + mt * 16 + g + h16 * 8;
                if (OUT_BF16)
                    ((u16*)Eout)[(size_t)row * D_EMB + col] = f2bf(acc[mt][nt][g]);
                else
                    ((float*)Eout)[(size_t)row * D_EMB + col] = acc[mt][nt][g];
            }
        }
}

// ---------------------------------------------------------------------------
// Kernel 2: per-class mean prototypes.  grid (N_WAY, D_EMB/256), 256 threads.
// ---------------------------------------------------------------------------
__global__ __launch_bounds__(256) void proto_kernel(const float* __restrict__ Es,
                                                    const int* __restrict__ labels,
                                                    float* __restrict__ proto) {
    __shared__ int labs[N_SUP];
    const int tid = threadIdx.x;
#pragma unroll
    for (int j = 0; j < N_SUP / 256; ++j) labs[j * 256 + tid] = labels[j * 256 + tid];
    __syncthreads();

    const int c   = blockIdx.x;
    const int col = blockIdx.y * 256 + tid;
    float s = 0.f;
    int cnt = 0;
    for (int r = 0; r < N_SUP; ++r) {
        if (labs[r] == c) { s += Es[(size_t)r * D_EMB + col]; ++cnt; }
    }
    proto[(size_t)c * D_EMB + col] = s / (float)cnt;
}

// ---------------------------------------------------------------------------
// Kernel 2b: p_sq[c] = ||proto[c]||^2.  grid N_WAY, 256 threads.
// ---------------------------------------------------------------------------
__global__ __launch_bounds__(256) void psq_kernel(const float* __restrict__ proto,
                                                  float* __restrict__ psq) {
    __shared__ float red[256];
    const int c = blockIdx.x, tid = threadIdx.x;
    float s = 0.f;
#pragma unroll
    for (int j = 0; j < D_EMB / 256; ++j) {
        const float v = proto[(size_t)c * D_EMB + j * 256 + tid];
        s += v * v;
    }
    red[tid] = s;
    __syncthreads();
    for (int off = 128; off > 0; off >>= 1) {
        if (tid < off) red[tid] += red[tid + off];
        __syncthreads();
    }
    if (tid == 0) psq[c] = red[0];
}

// ---------------------------------------------------------------------------
// Kernel 3: out = -sqrt(max(q_sq + p_sq - 2 * (Eq @ protoT), 0))
// Block: 128 queries x all 64 prototypes; 8 waves, each wave 16 rows x 64 cols.
// Q tile is a raw bf16 copy -> CDNA5 async global->LDS (ASYNCcnt-tracked).
// q_sq is accumulated in-flight from the same bf16 A fragments.
// ---------------------------------------------------------------------------
__global__ __launch_bounds__(256) void dist_kernel(const u16* __restrict__ Eq,
                                                   const float* __restrict__ proto,
                                                   const float* __restrict__ psq,
                                                   float* __restrict__ out) {
    constexpr int AS = 40, BS = 40;
    __shared__ __align__(16) u16 sQ[128 * AS];
    __shared__ __align__(16) u16 sP[64 * BS];

    const int tid  = threadIdx.x;
    const int lane = tid & 31;
    const int w    = tid >> 5;
    const int l16  = lane & 15;
    const int h16  = lane >> 4;
    const int bm0  = blockIdx.x * 128;

    f32x8 acc[4];
#pragma unroll
    for (int nt = 0; nt < 4; ++nt)
#pragma unroll
        for (int g = 0; g < 8; ++g) acc[nt][g] = 0.f;
    float qsq = 0.f;

    for (int kk = 0; kk < D_EMB; kk += 32) {
        // --- Q tile: already bf16 in global; 16B per lane per issue ---
#pragma unroll
        for (int j = 0; j < 2; ++j) {
            const int e = j * 256 + tid;          // 0..511 8-elem groups
            const int r = e >> 2, c8 = (e & 3) << 3;
#if HAVE_ASYNC_LDS
            __builtin_amdgcn_global_load_async_to_lds_b128(
                (glb_v4i*)(Eq + (size_t)(bm0 + r) * D_EMB + kk + c8),
                (lds_v4i*)&sQ[r * AS + c8], 0, 0);
#else
            *(u16x8*)&sQ[r * AS + c8] =
                *(const u16x8*)(Eq + (size_t)(bm0 + r) * D_EMB + kk + c8);
#endif
        }
        // --- P tile: proto row-major [n][k] already matches B=[col][k] ---
        // (VALU conversion overlaps the async Q-tile copy)
#pragma unroll
        for (int j = 0; j < 2; ++j) {
            const int e = j * 256 + tid;          // 0..511 float4s
            const int p = e >> 3, c4 = (e & 7) << 2;
            const float4 v = *(const float4*)(proto + (size_t)p * D_EMB + kk + c4);
            const int o = p * BS + c4;
            sP[o] = f2bf(v.x); sP[o + 1] = f2bf(v.y);
            sP[o + 2] = f2bf(v.z); sP[o + 3] = f2bf(v.w);
        }
        if (kk + 32 < D_EMB && tid < 64)
            __builtin_prefetch(proto + (size_t)tid * D_EMB + (kk + 32), 0, 0);
#if HAVE_ASYNC_LDS
        WAIT_ASYNC();
#endif
        __syncthreads();

        const int ka = h16 << 3;
        const int ro = (w * 16 + l16) * AS + ka;
        const u16x8 alo = *(const u16x8*)&sQ[ro];
        const u16x8 ahi = *(const u16x8*)&sQ[ro + 16];
#pragma unroll
        for (int i = 0; i < 8; ++i) {
            const float f0 = bf2f(alo[i]); qsq += f0 * f0;
            const float f1 = bf2f(ahi[i]); qsq += f1 * f1;
        }
        const bf16x16 afr = mk_frag(alo, ahi);
        const int kb = h16 << 3;
#pragma unroll
        for (int nt = 0; nt < 4; ++nt) {
            const int co = (nt * 16 + l16) * BS + kb;
            const bf16x16 bfr = mk_frag(*(const u16x8*)&sP[co], *(const u16x8*)&sP[co + 16]);
            acc[nt] = wmma_bf16(afr, bfr, acc[nt]);
        }
        __syncthreads();
    }

    // lane L and L^16 together hold row (L&15); reduce across halves
    const float qrow = qsq + __shfl_xor(qsq, 16);
    const int roff = h16 * 8;
#pragma unroll
    for (int nt = 0; nt < 4; ++nt) {
        const float ps = psq[nt * 16 + l16];
#pragma unroll
        for (int g = 0; g < 8; ++g) {
            const float q  = __shfl(qrow, g + roff);   // q_sq of this C-row
            const float sq = fmaxf(q + ps - 2.0f * acc[nt][g], 0.0f);
            out[(size_t)(bm0 + w * 16 + g + roff) * N_WAY + nt * 16 + l16] = -sqrtf(sq);
        }
    }
}

// ---------------------------------------------------------------------------
// Launch. Inputs: support_set, query_set, support_labels, n_way, W, b
// ---------------------------------------------------------------------------
extern "C" void kernel_launch(void* const* d_in, const int* in_sizes, int n_in,
                              void* d_out, int out_size, void* d_ws, size_t ws_size,
                              hipStream_t stream) {
    const float* sup    = (const float*)d_in[0];
    const float* qry    = (const float*)d_in[1];
    const int*   labels = (const int*)d_in[2];
    const float* W      = (const float*)d_in[4];
    const float* bias   = (const float*)d_in[5];
    float*       out    = (float*)d_out;

    // workspace: Eq (bf16, 32MB) | Es (fp32, 8MB) | proto (fp32) | psq (fp32)
    u16*   Eq    = (u16*)d_ws;
    float* Es    = (float*)((char*)d_ws + (size_t)N_QRY * D_EMB * sizeof(u16));
    float* proto = Es + (size_t)N_SUP * D_EMB;
    float* psq   = proto + (size_t)N_WAY * D_EMB;

    const dim3 blk(256);
    embed_kernel<true ><<<dim3(D_EMB / 64, N_QRY / 128), blk, 0, stream>>>(qry, W, bias, (void*)Eq);
    embed_kernel<false><<<dim3(D_EMB / 64, N_SUP / 128), blk, 0, stream>>>(sup, W, bias, (void*)Es);
    proto_kernel<<<dim3(N_WAY, D_EMB / 256), blk, 0, stream>>>(Es, labels, proto);
    psq_kernel<<<N_WAY, blk, 0, stream>>>(proto, psq);
    dist_kernel<<<N_QRY / 128, blk, 0, stream>>>(Eq, proto, psq, out);

    (void)in_sizes; (void)n_in; (void)out_size; (void)ws_size;
}
